// SpikingUnifiedCoreFlow_61589831025171
// MI455X (gfx1250) — matrix operand order repaired and checked by reading, not compiled, not run on hardware
//
#include <hip/hip_runtime.h>
#include <math.h>

// ---------------------------------------------------------------------------
// SpikingUnifiedCoreFlow for MI455X (gfx1250, wave32, WMMA).
//
// Per cycle: spikes(binary) @ W[l] for l=0..3 with threshold/fire/reset fused
// into the GEMM epilogue. Weights converted once to bf16 (Wt = W^T), 128 MB
// total -> resident in the 192 MB L2 across all 32 cycles. Math runs on
// v_wmma_f32_16x16x32_bf16 with f32 accumulation of membrane potentials.
//
// GEMM tiling (v3): 2x2 register blocking per wave (32Mx32N), 4 waves per
// block arranged 2Mx2N (block tile 64Mx64N), grid = 256 blocks. K-loop is
// explicitly double-buffered (two independent fragment register groups,
// unroll x2) so loads for step k+1 are in flight while WMMAs consume step k,
// avoiding full s_wait_loadcnt 0x0 drains between WMMA pairs.
//
// Workspace layout (requires ~152 MB):
//   [0)      Wt   : 4 * 4096*4096 bf16 (W transposed, K contiguous)
//   [+128MB) pot  : 4 * 256*4096  f32
//   [+16MB)  cnt  : 256*4096      f32
//   [+4MB)   spk0 : 256*4096      bf16
//   [+2MB)   spk1 : 256*4096      bf16
// ---------------------------------------------------------------------------

typedef __bf16 bf16;
typedef __attribute__((ext_vector_type(16))) __bf16 v16bf;
typedef __attribute__((ext_vector_type(8)))  float  v8f;
typedef __attribute__((ext_vector_type(4)))  unsigned int v4u;

#define DD   4096
#define BB   256
#define TT   32
#define NLAY 4

// ---------------------------------------------------------------------------
// W (f32 row-major, [K=D][N=D]) -> Wt (bf16, [N=D][K=D]) transpose+convert.
// LDS-tiled 32x32 so both global read and write are coalesced. One-time cost.
__global__ __launch_bounds__(256) void wprep_kernel(
    const float* __restrict__ W0, const float* __restrict__ W1,
    const float* __restrict__ W2, const float* __restrict__ W3,
    bf16* __restrict__ Wt)
{
    const float* W;
    switch (blockIdx.z) {
        case 0: W = W0; break;
        case 1: W = W1; break;
        case 2: W = W2; break;
        default: W = W3; break;
    }
    bf16* out = Wt + (size_t)blockIdx.z * DD * DD;

    __shared__ float tile[32][33];
    const int n0 = blockIdx.x * 32;
    const int k0 = blockIdx.y * 32;
    const int tx = threadIdx.x;        // 0..31
    const int ty = threadIdx.y;        // 0..7

    #pragma unroll
    for (int i = 0; i < 32; i += 8)
        tile[ty + i][tx] = W[(size_t)(k0 + ty + i) * DD + (n0 + tx)];
    __syncthreads();
    #pragma unroll
    for (int i = 0; i < 32; i += 8)
        out[(size_t)(n0 + ty + i) * DD + (k0 + tx)] = (bf16)tile[tx][ty + i];
}

// ---------------------------------------------------------------------------
__global__ __launch_bounds__(256) void zero_kernel(float* __restrict__ p, int n)
{
    int i = blockIdx.x * blockDim.x + threadIdx.x;
    if (i < n) p[i] = 0.0f;
}

// ---------------------------------------------------------------------------
// Faithful port of to_uniform_spikes for one cycle; emits bf16 0/1.
__global__ __launch_bounds__(256) void spike_kernel(
    const float* __restrict__ x, int cycle, bf16* __restrict__ spk)
{
    int i = blockIdx.x * blockDim.x + threadIdx.x;
    float r = x[i];
    float Nf = roundf(r * 32.0f);
    int   N  = (int)Nf;
    float outv;
    if (N == 32) {
        outv = 1.0f;
    } else if (N == 0) {
        outv = 0.0f;
    } else {
        float spacing = 32.0f / Nf;
        float c = (float)cycle;
        bool fire = (floorf(c / spacing) < Nf) &&
                    (floorf(fmodf(c, spacing)) == 0.0f);
        outv = fire ? 1.0f : 0.0f;
    }
    spk[i] = (bf16)outv;
}

// ---------------------------------------------------------------------------
// Fused GEMM + integrate + fire + reset for one (cycle, layer).
//   delta = Sin[256xD] @ W[DxD]   (bf16 WMMA, f32 acc)
//   pot  += delta; fire = pot >= thr; pot -= fire*thr; Sout = fire (bf16)
//   if addCounts: cnt += fire
//
// Block = 128 threads (4 waves) arranged 2(M) x 2(N); each wave owns a
// 32Mx32N tile = 2x2 accumulators. Block tile = 64Mx64N.
// Grid = (D/64, 256/64) = (64, 4) = 256 blocks.
//
// Fragment layouts per CDNA5 ISA 7.12.2 (wave32):
//   A 16x32 bf16 : lane&15 = row M; lanes 0-15 hold K [k..k+7],[k+16..k+23],
//                  lanes 16-31 hold K [k+8..k+15],[k+24..k+31].
//   B 32x16 bf16 : lane&15 = col N; lanes 0-15 hold K [k..k+7],[k+16..k+23],
//                  lanes 16-31 hold K [k+8..k+15],[k+24..k+31] (from Wt, K-contiguous).
//   C/D 16x16 f32: lane&15 = col N; VGPR r = row M0 + r + 8*(lane>>4).
__global__ __launch_bounds__(128) void gemm_fire_kernel(
    const bf16* __restrict__ Sin,     // [256][D] bf16 spikes in
    const bf16* __restrict__ Wt,      // [D][D]  bf16, W transposed (N-major)
    float*      __restrict__ pot,     // [256][D] f32 membrane potential
    const float* __restrict__ thrp,   // thresholds[4]
    int layer,
    bf16*       __restrict__ Sout,    // [256][D] bf16 spikes out
    float*      __restrict__ cnt,     // [256][D] f32 spike counts
    int addCounts)
{
    const int tid    = threadIdx.x;
    const int wave   = tid >> 5;
    const int lane   = tid & 31;
    const int laneLo = lane & 15;
    const int laneHi = lane >> 4;

    const int waveM0 = blockIdx.y * 64 + (wave >> 1) * 32;   // 2 M-tiles
    const int waveN0 = blockIdx.x * 64 + (wave & 1) * 32;    // 2 N-tiles

    const float thr = thrp[layer];

    // A: one spike row per lane per m-tile; chunks at +laneHi*8 and +16.
    const bf16* arow[2];
    #pragma unroll
    for (int mt = 0; mt < 2; ++mt)
        arow[mt] = Sin + (size_t)(waveM0 + mt * 16 + laneLo) * DD + laneHi * 8;

    // B: one Wt row (= one W column) per lane per n-tile, K-contiguous.
    const bf16* brow[2];
    #pragma unroll
    for (int nt = 0; nt < 2; ++nt)
        brow[nt] = Wt + (size_t)(waveN0 + nt * 16 + laneLo) * DD + laneHi * 8;

    union Frag { v16bf v; v4u q[2]; };
    Frag aP[2], bP[2], aQ[2], bQ[2];   // ping (P) / pong (Q) register groups
    v8f acc[2][2] = {};

    // Prologue: group P <- K = 0.
    #pragma unroll
    for (int mt = 0; mt < 2; ++mt) {
        aP[mt].q[0] = *(const v4u*)(arow[mt]);
        aP[mt].q[1] = *(const v4u*)(arow[mt] + 16);
    }
    #pragma unroll
    for (int nt = 0; nt < 2; ++nt) {
        bP[nt].q[0] = *(const v4u*)(brow[nt]);
        bP[nt].q[1] = *(const v4u*)(brow[nt] + 16);
    }

    for (int kk = 0; kk < DD; kk += 64) {
        const int k1 = kk + 32;               // always < DD
        const int k2 = (kk + 64) & (DD - 1);  // wraps to 0 on last iter (junk, unused)

        // Issue group-Q loads (K = k1) before consuming group-P.
        #pragma unroll
        for (int mt = 0; mt < 2; ++mt) {
            aQ[mt].q[0] = *(const v4u*)(arow[mt] + k1);
            aQ[mt].q[1] = *(const v4u*)(arow[mt] + k1 + 16);
            __builtin_prefetch(arow[mt] + k1 + 96, 0, 1);
        }
        #pragma unroll
        for (int nt = 0; nt < 2; ++nt) {
            bQ[nt].q[0] = *(const v4u*)(brow[nt] + k1);
            bQ[nt].q[1] = *(const v4u*)(brow[nt] + k1 + 16);
            __builtin_prefetch(brow[nt] + k1 + 96, 0, 1);
        }
        #pragma unroll
        for (int mt = 0; mt < 2; ++mt)
            #pragma unroll
            for (int nt = 0; nt < 2; ++nt)
                acc[mt][nt] = __builtin_amdgcn_wmma_f32_16x16x32_bf16(
                    false, aP[mt].v, false, bP[nt].v, (short)0, acc[mt][nt],
                    false, false);

        // Issue group-P loads (K = k2) before consuming group-Q.
        #pragma unroll
        for (int mt = 0; mt < 2; ++mt) {
            aP[mt].q[0] = *(const v4u*)(arow[mt] + k2);
            aP[mt].q[1] = *(const v4u*)(arow[mt] + k2 + 16);
            __builtin_prefetch(arow[mt] + k2 + 96, 0, 1);
        }
        #pragma unroll
        for (int nt = 0; nt < 2; ++nt) {
            bP[nt].q[0] = *(const v4u*)(brow[nt] + k2);
            bP[nt].q[1] = *(const v4u*)(brow[nt] + k2 + 16);
            __builtin_prefetch(brow[nt] + k2 + 96, 0, 1);
        }
        #pragma unroll
        for (int mt = 0; mt < 2; ++mt)
            #pragma unroll
            for (int nt = 0; nt < 2; ++nt)
                acc[mt][nt] = __builtin_amdgcn_wmma_f32_16x16x32_bf16(
                    false, aQ[mt].v, false, bQ[nt].v, (short)0, acc[mt][nt],
                    false, false);
    }

    // Fused epilogue: integrate, fire, subtract-reset, emit next spikes.
    #pragma unroll
    for (int mt = 0; mt < 2; ++mt) {
        #pragma unroll
        for (int nt = 0; nt < 2; ++nt) {
            const int col = waveN0 + nt * 16 + laneLo;
            #pragma unroll
            for (int r = 0; r < 8; ++r) {
                const int row = waveM0 + mt * 16 + r + 8 * laneHi;
                const size_t idx = (size_t)row * DD + col;
                float p = pot[idx] + acc[mt][nt][r];
                float fire = (p >= thr) ? 1.0f : 0.0f;
                p -= fire * thr;
                pot[idx]  = p;
                Sout[idx] = (bf16)fire;
                if (addCounts) cnt[idx] += fire;
            }
        }
    }
}

// ---------------------------------------------------------------------------
__global__ __launch_bounds__(256) void finalize_kernel(
    const float* __restrict__ cnt, float* __restrict__ out)
{
    int i = blockIdx.x * blockDim.x + threadIdx.x;
    out[i] = cnt[i] * (1.0f / 32.0f);
}

// ---------------------------------------------------------------------------
extern "C" void kernel_launch(void* const* d_in, const int* in_sizes, int n_in,
                              void* d_out, int out_size, void* d_ws, size_t ws_size,
                              hipStream_t stream)
{
    const float* x   = (const float*)d_in[0];
    const float* W0  = (const float*)d_in[1];
    const float* W1  = (const float*)d_in[2];
    const float* W2  = (const float*)d_in[3];
    const float* W3  = (const float*)d_in[4];
    const float* thr = (const float*)d_in[5];
    float* out = (float*)d_out;

    char* ws = (char*)d_ws;
    bf16*  Wt   = (bf16*)ws;                                            // 128 MB
    float* pot  = (float*)(ws + (size_t)NLAY * DD * DD * sizeof(bf16)); // 16 MB
    float* cnt  = pot + (size_t)NLAY * BB * DD;                         // 4 MB
    bf16*  spk0 = (bf16*)(cnt + (size_t)BB * DD);                       // 2 MB
    bf16*  spk1 = spk0 + (size_t)BB * DD;                               // 2 MB

    // One-time (per call) weight transpose+convert: f32 [K][N] -> bf16 [N][K].
    wprep_kernel<<<dim3(DD / 32, DD / 32, NLAY), dim3(32, 8), 0, stream>>>(
        W0, W1, W2, W3, Wt);

    // Zero potentials (4*B*D) and counts (B*D) — contiguous in ws.
    {
        int n = (NLAY + 1) * BB * DD;
        zero_kernel<<<(n + 255) / 256, 256, 0, stream>>>(pot, n);
    }

    for (int c = 0; c < TT; ++c) {
        spike_kernel<<<(BB * DD) / 256, 256, 0, stream>>>(x, c, spk0);
        for (int l = 0; l < NLAY; ++l) {
            const bf16* Sin  = (l & 1) ? spk1 : spk0;
            bf16*       So   = (l & 1) ? spk0 : spk1;
            gemm_fire_kernel<<<dim3(DD / 64, BB / 64), 128, 0, stream>>>(
                Sin, Wt + (size_t)l * DD * DD, pot + (size_t)l * BB * DD,
                thr, l, So, cnt, (l == NLAY - 1) ? 1 : 0);
        }
    }

    finalize_kernel<<<(BB * DD) / 256, 256, 0, stream>>>(cnt, out);
}
